// CifNetSelfAttention_7258494730484
// MI455X (gfx1250) — matrix operand order repaired and checked by reading, not compiled, not run on hardware
//
#include <hip/hip_runtime.h>
#include <hip/hip_bf16.h>

// ---------------------------------------------------------------------------
// CifNet self-attention for MI455X (gfx1250, wave32, WMMA).
// f32->bf16 cvt -> 3x implicit-GEMM conv (QKV, Q pre-scaled by 1/sqrt(64))
//   -> flash attention (64-key blocks, row-sum via WMMA ones-trick)
//   -> implicit-GEMM conv (output, f32 NCHW).
// All matrix math via v_wmma_f32_16x16x32_bf16 (f32 accumulate).
// ---------------------------------------------------------------------------

typedef unsigned short u16;
typedef __bf16 bf16_t;
typedef bf16_t v16bf __attribute__((ext_vector_type(16)));
typedef float  v8f   __attribute__((ext_vector_type(8)));

#define HID    256
#define INTERC 512
#define HEADS  8
#define CH     64
#define HWDIM  48
#define P2     (HWDIM * HWDIM)   // 2304 spatial positions per image
#define BATCH  4
#define NTOT   (BATCH * P2)      // 9216

#define WMMA_BF16(A, B, C) \
  __builtin_amdgcn_wmma_f32_16x16x32_bf16(false, (A), false, (B), (short)0, (C), false, false)

// round-to-nearest-even f32 -> bf16 (as raw u16)
static __device__ __forceinline__ u16 f2bf(float f) {
  unsigned u = __float_as_uint(f);
  u += 0x7FFFu + ((u >> 16) & 1u);
  return (u16)(u >> 16);
}

union FragU {
  struct { uint4 lo, hi; } q;
  v16bf v;
};

// A-operand (16x32 bf16) fragment, row-major source [row][K], K contiguous.
// ISA layout: lanes 0-15: halves 0-7 = K0..7, halves 8-15 = K16..23
//             lanes 16-31: halves 0-7 = K8..15, halves 8-15 = K24..31
static __device__ __forceinline__ v16bf load_a_frag(const u16* base, int row_stride) {
  const int lane = threadIdx.x & 31;
  const u16* p = base + (lane & 15) * row_stride + ((lane < 16) ? 0 : 8);
  FragU f;
  f.q.lo = *(const uint4*)(p);        // 8 halves
  f.q.hi = *(const uint4*)(p + 16);   // 8 halves
  return f.v;
}

// B-operand (32x16 bf16) fragment, source laid out [col][K] with K contiguous.
// Layout: col = lane&15; lanes 0-15 hold K0..7,K16..23, lanes 16-31 hold K8..15,K24..31.
static __device__ __forceinline__ v16bf load_b_frag(const u16* base, int col_stride) {
  const int lane = threadIdx.x & 31;
  const u16* p = base + (lane & 15) * col_stride + ((lane < 16) ? 0 : 8);
  FragU f;
  f.q.lo = *(const uint4*)(p);
  f.q.hi = *(const uint4*)(p + 16);
  return f.v;
}

// constant bf16(1.0) fragment (layout-independent)
static __device__ __forceinline__ v16bf ones_frag() {
  FragU f;
  const unsigned o = 0x3F803F80u;
  f.q.lo = make_uint4(o, o, o, o);
  f.q.hi = make_uint4(o, o, o, o);
  return f.v;
}

// ---------------------------------------------------------------------------
// fp32 -> bf16 conversion (grid-stride)
// ---------------------------------------------------------------------------
__global__ __launch_bounds__(256)
void cvt_f32_bf16(const float* __restrict__ s, u16* __restrict__ d, int n) {
  int i = blockIdx.x * 256 + threadIdx.x;
  const int stride = gridDim.x * 256;
  for (; i < n; i += stride) d[i] = f2bf(s[i]);
}

// ---------------------------------------------------------------------------
// 3x3 SAME conv as implicit GEMM:  O[co][n] = oscale * sum_k W[co][k]*im2col[k][n]
//   M = Cout, N = BATCH*P2, K = Cin*9
// Block: 256 threads = 8 waves; block tile 64(co) x 128(sp);
// each wave a 32x32 macro-tile (4 wmmas per K-step, A/B fragment reuse).
// SRC_NCHW: 1 -> src bf16 [b][Cin][48][48]; 0 -> src bf16 [b][hw][Cin]
// DST_F32 : 0 -> dst bf16 [b*hw][Cout];     1 -> dst f32 NCHW [b][Cout][hw]
// ---------------------------------------------------------------------------
template<int SRC_NCHW, int DST_F32>
__global__ __launch_bounds__(256)
void conv3x3_wmma(const u16* __restrict__ src, const u16* __restrict__ wgt,
                  int Cin, int Cout, float oscale,
                  u16* __restrict__ dst_bf, float* __restrict__ dst_f) {
  __shared__ __align__(16) u16 Bs[128 * 32];   // [sp_local][k_local], K contiguous
  const int K = Cin * 9;
  const int tid  = threadIdx.x;
  const int lane = tid & 31;
  const int wave = tid >> 5;
  const int co0    = blockIdx.x * 64 + (wave & 1) * 32;
  const int spl0   = (wave >> 1) * 32;
  const int spBase = blockIdx.y * 128;

  // staging assignment: thread owns sp row (tid>>1), k halves (tid&1)*16..+15.
  // spatial decode hoisted out of the K loop.
  const int s_spl = tid >> 1;
  const int s_kl0 = (tid & 1) * 16;
  const int s_n   = spBase + s_spl;
  const int s_bb  = s_n / P2;
  const int s_pos = s_n - s_bb * P2;
  const int s_y   = s_pos / HWDIM, s_x = s_pos - s_y * HWDIM;

  v8f acc00 = {}, acc01 = {}, acc10 = {}, acc11 = {};
  for (int k0 = 0; k0 < K; k0 += 32) {
    __syncthreads();                      // protect previous iteration's reads
    #pragma unroll
    for (int i = 0; i < 16; ++i) {
      const int k  = k0 + s_kl0 + i;
      const int ci = k / 9, r9 = k - ci * 9;
      const int yy = s_y + r9 / 3 - 1, xx = s_x + (r9 % 3) - 1;
      u16 val = 0;
      if (yy >= 0 && yy < HWDIM && xx >= 0 && xx < HWDIM) {
        val = SRC_NCHW ? src[((size_t)(s_bb * Cin + ci) * HWDIM + yy) * HWDIM + xx]
                       : src[(size_t)(s_bb * P2 + yy * HWDIM + xx) * Cin + ci];
      }
      Bs[s_spl * 32 + s_kl0 + i] = val;
    }
    __syncthreads();

    const u16* wrow = wgt + (size_t)co0 * K + k0;
    if (k0 + 32 < K) __builtin_prefetch(wrow + 32, 0, 3);   // global_prefetch_b8
    const v16bf a0 = load_a_frag(wrow, K);
    const v16bf a1 = load_a_frag(wrow + (size_t)16 * K, K);
    const v16bf b0 = load_b_frag(Bs + spl0 * 32, 32);
    const v16bf b1 = load_b_frag(Bs + (spl0 + 16) * 32, 32);
    acc00 = WMMA_BF16(a0, b0, acc00);
    acc01 = WMMA_BF16(a0, b1, acc01);
    acc10 = WMMA_BF16(a1, b0, acc10);
    acc11 = WMMA_BF16(a1, b1, acc11);
  }

  // epilogue: D layout VGPR r <-> row (co) r + 8*hi, col (sp) = lane&15
  const int hiOff = (lane < 16) ? 0 : 8;
  #pragma unroll
  for (int ts = 0; ts < 2; ++ts) {
    const int n  = spBase + spl0 + ts * 16 + (lane & 15);
    const int bb = n / P2, pos = n - bb * P2;
    #pragma unroll
    for (int tc = 0; tc < 2; ++tc) {
      const v8f acc = (tc == 0) ? (ts == 0 ? acc00 : acc01)
                                : (ts == 0 ? acc10 : acc11);
      #pragma unroll
      for (int r = 0; r < 8; ++r) {
        const int co = co0 + tc * 16 + r + hiOff;
        const float v = acc[r] * oscale;
        if (DST_F32) dst_f[((size_t)(bb * Cout + co)) * P2 + pos] = v;
        else         dst_bf[(size_t)n * Cout + co] = f2bf(v);
      }
    }
  }
}

// ---------------------------------------------------------------------------
// Flash attention: per (b, head), Q[2304x64] K^T -> softmax -> *V.
// Q was pre-scaled by 1/sqrt(64) in the conv epilogue.
// Block = 128 threads (4 waves); each wave owns a 16-row Q tile, streams over
// 36 key blocks of 64. Row sums come from WMMA against a ones fragment.
// ---------------------------------------------------------------------------
__global__ __launch_bounds__(128)
void flash_attn_wmma(const u16* __restrict__ q, const u16* __restrict__ k,
                     const u16* __restrict__ v, u16* __restrict__ o) {
  __shared__ __align__(16) u16 Kb[64 * 64];      // [key][c]   (c contiguous)
  __shared__ __align__(16) u16 Vt[64 * 64];      // [c][key]   (key contiguous)
  __shared__ __align__(16) u16 Ps[4 * 16 * 64];  // per-wave P [row][key]

  const int tid = threadIdx.x, lane = tid & 31, wave = tid >> 5;
  const int bh = blockIdx.y;               // 0..31
  const int b = bh >> 3, head = bh & 7;
  const int q0 = (blockIdx.x * 4 + wave) * 16;
  const size_t rowOff = (size_t)(b * P2) * INTERC + head * CH;

  // Q fragments held in VGPRs for the whole loop (c 0..31 and 32..63)
  const u16* qbase = q + rowOff + (size_t)q0 * INTERC;
  const v16bf qa0 = load_a_frag(qbase, INTERC);
  const v16bf qa1 = load_a_frag(qbase + 32, INTERC);
  const v16bf onesB = ones_frag();

  v8f acc0 = {}, acc1 = {}, acc2 = {}, acc3 = {};
  v8f lacc = {};                           // row sums (replicated across cols)
  float m[8];
  #pragma unroll
  for (int r = 0; r < 8; ++r) m[r] = -3.0e38f;
  u16* myPs = Ps + wave * 16 * 64;

  for (int kb0 = 0; kb0 < P2; kb0 += 64) {
    __syncthreads();
    // ---- stage K [64][64] (b128 copies) and V transposed [64][64] ----
    for (int ch = tid; ch < 512; ch += 128) {
      const int key = ch >> 3, c8 = (ch & 7) * 8;
      *(uint4*)(Kb + key * 64 + c8) =
          *(const uint4*)(k + rowOff + (size_t)(kb0 + key) * INTERC + c8);
    }
    for (int idx = tid; idx < 4096; idx += 128) {
      const int key = idx >> 6, c = idx & 63;
      Vt[c * 64 + key] = v[rowOff + (size_t)(kb0 + key) * INTERC + c];
    }
    if (kb0 + 64 < P2) {                   // prefetch next key block
      const int key = kb0 + 64 + (tid & 63);
      const u16* pf = (tid < 64 ? k : v) + rowOff + (size_t)key * INTERC;
      __builtin_prefetch(pf, 0, 3);
    }
    __syncthreads();

    // ---- S tiles: 4 groups of 16 keys, K-dim 64 = 2 wmmas each ----
    v8f s0 = {}, s1 = {}, s2 = {}, s3 = {};
    s0 = WMMA_BF16(qa0, load_b_frag(Kb + 0 * 16 * 64, 64), s0);
    s0 = WMMA_BF16(qa1, load_b_frag(Kb + 0 * 16 * 64 + 32, 64), s0);
    s1 = WMMA_BF16(qa0, load_b_frag(Kb + 1 * 16 * 64, 64), s1);
    s1 = WMMA_BF16(qa1, load_b_frag(Kb + 1 * 16 * 64 + 32, 64), s1);
    s2 = WMMA_BF16(qa0, load_b_frag(Kb + 2 * 16 * 64, 64), s2);
    s2 = WMMA_BF16(qa1, load_b_frag(Kb + 2 * 16 * 64 + 32, 64), s2);
    s3 = WMMA_BF16(qa0, load_b_frag(Kb + 3 * 16 * 64, 64), s3);
    s3 = WMMA_BF16(qa1, load_b_frag(Kb + 3 * 16 * 64 + 32, 64), s3);

    // ---- online softmax (max butterfly only; sums via WMMA below) ----
    #pragma unroll
    for (int r = 0; r < 8; ++r) {
      const float a0 = s0[r], a1 = s1[r], a2 = s2[r], a3 = s3[r];
      float rmax = fmaxf(fmaxf(a0, a1), fmaxf(a2, a3));
      #pragma unroll
      for (int off = 1; off < 16; off <<= 1)
        rmax = fmaxf(rmax, __shfl_xor(rmax, off, 32));
      const float mn   = fmaxf(m[r], rmax);
      const float corr = __expf(m[r] - mn);
      m[r] = mn;
      acc0[r] *= corr; acc1[r] *= corr; acc2[r] *= corr; acc3[r] *= corr;
      lacc[r] *= corr;
      const int row = r + ((lane < 16) ? 0 : 8);
      myPs[row * 64 +  0 + (lane & 15)] = f2bf(__expf(a0 - mn));
      myPs[row * 64 + 16 + (lane & 15)] = f2bf(__expf(a1 - mn));
      myPs[row * 64 + 32 + (lane & 15)] = f2bf(__expf(a2 - mn));
      myPs[row * 64 + 48 + (lane & 15)] = f2bf(__expf(a3 - mn));
    }

    // ---- P fragments (keys 0..31, 32..63); row sums l += P * ones ----
    const v16bf pa0 = load_a_frag(myPs, 64);
    const v16bf pa1 = load_a_frag(myPs + 32, 64);
    lacc = WMMA_BF16(pa0, onesB, lacc);
    lacc = WMMA_BF16(pa1, onesB, lacc);

    // ---- acc += P * V  (4 c-tiles of 16, K-dim 64 = 2 wmmas each) ----
    acc0 = WMMA_BF16(pa0, load_b_frag(Vt + 0 * 16 * 64, 64), acc0);
    acc0 = WMMA_BF16(pa1, load_b_frag(Vt + 0 * 16 * 64 + 32, 64), acc0);
    acc1 = WMMA_BF16(pa0, load_b_frag(Vt + 1 * 16 * 64, 64), acc1);
    acc1 = WMMA_BF16(pa1, load_b_frag(Vt + 1 * 16 * 64 + 32, 64), acc1);
    acc2 = WMMA_BF16(pa0, load_b_frag(Vt + 2 * 16 * 64, 64), acc2);
    acc2 = WMMA_BF16(pa1, load_b_frag(Vt + 2 * 16 * 64 + 32, 64), acc2);
    acc3 = WMMA_BF16(pa0, load_b_frag(Vt + 3 * 16 * 64, 64), acc3);
    acc3 = WMMA_BF16(pa1, load_b_frag(Vt + 3 * 16 * 64 + 32, 64), acc3);
  }

  // ---- normalize and write bf16 [b*hw][512] (per-head 64-ch slice) ----
  #pragma unroll
  for (int r = 0; r < 8; ++r) {
    const int row = q0 + r + ((lane < 16) ? 0 : 8);
    const float inv = 1.0f / lacc[r];
    const size_t base = (size_t)(b * P2 + row) * INTERC + head * CH + (lane & 15);
    o[base + 0]  = f2bf(acc0[r] * inv);
    o[base + 16] = f2bf(acc1[r] * inv);
    o[base + 32] = f2bf(acc2[r] * inv);
    o[base + 48] = f2bf(acc3[r] * inv);
  }
}

// ---------------------------------------------------------------------------
// Host-side launch
// ---------------------------------------------------------------------------
extern "C" void kernel_launch(void* const* d_in, const int* in_sizes, int n_in,
                              void* d_out, int out_size, void* d_ws, size_t ws_size,
                              hipStream_t stream) {
  (void)in_sizes; (void)n_in; (void)out_size; (void)ws_size;
  const float* x  = (const float*)d_in[0];
  const float* wq = (const float*)d_in[1];
  const float* wk = (const float*)d_in[2];
  const float* wv = (const float*)d_in[3];
  const float* wo = (const float*)d_in[4];
  float* out = (float*)d_out;

  // Workspace carve (~52 MB total): bf16 copies of x/weights + Q/K/V/attn.
  char* ws = (char*)d_ws;
  size_t off = 0;
  auto carve = [&](size_t elems) {
    u16* p = (u16*)(ws + off);
    off += ((elems * 2 + 255) & ~(size_t)255);
    return p;
  };
  u16* xb  = carve((size_t)BATCH * HID * P2);
  u16* wqb = carve((size_t)INTERC * HID * 9);
  u16* wkb = carve((size_t)INTERC * HID * 9);
  u16* wvb = carve((size_t)INTERC * HID * 9);
  u16* wob = carve((size_t)HID * INTERC * 9);
  u16* qb  = carve((size_t)BATCH * P2 * INTERC);
  u16* kb  = carve((size_t)BATCH * P2 * INTERC);
  u16* vb  = carve((size_t)BATCH * P2 * INTERC);
  u16* ab  = carve((size_t)BATCH * P2 * INTERC);

  auto cvt = [&](const float* s, u16* d, int n) {
    int blocks = (n + 255) / 256; if (blocks > 2048) blocks = 2048;
    hipLaunchKernelGGL(cvt_f32_bf16, dim3(blocks), dim3(256), 0, stream, s, d, n);
  };
  cvt(x,  xb,  BATCH * HID * P2);
  cvt(wq, wqb, INTERC * HID * 9);
  cvt(wk, wkb, INTERC * HID * 9);
  cvt(wv, wvb, INTERC * HID * 9);
  cvt(wo, wob, HID * INTERC * 9);

  // QKV convs: M=512, N=9216, K=2304; Q pre-scaled by 1/sqrt(64)
  const dim3 cgrid(INTERC / 64, NTOT / 128);
  hipLaunchKernelGGL((conv3x3_wmma<1, 0>), cgrid, dim3(256), 0, stream,
                     xb, wqb, HID, INTERC, 0.125f, qb, (float*)nullptr);
  hipLaunchKernelGGL((conv3x3_wmma<1, 0>), cgrid, dim3(256), 0, stream,
                     xb, wkb, HID, INTERC, 1.0f, kb, (float*)nullptr);
  hipLaunchKernelGGL((conv3x3_wmma<1, 0>), cgrid, dim3(256), 0, stream,
                     xb, wvb, HID, INTERC, 1.0f, vb, (float*)nullptr);

  // Flash attention: 144 q-tiles / 4 per block = 36 blocks, 32 (b,head) pairs
  hipLaunchKernelGGL(flash_attn_wmma, dim3(P2 / 64, BATCH * HEADS), dim3(128), 0, stream,
                     qb, kb, vb, ab);

  // Output conv: M=256, N=9216, K=4608, f32 NCHW epilogue into d_out
  const dim3 ogrid(HID / 64, NTOT / 128);
  hipLaunchKernelGGL((conv3x3_wmma<0, 1>), ogrid, dim3(256), 0, stream,
                     ab, wob, INTERC, HID, 1.0f, (u16*)nullptr, out);
}
